// GNNBaseLayer_86500641341823
// MI455X (gfx1250) — matrix operand order repaired and checked by reading, not compiled, not run on hardware
//
#include <hip/hip_runtime.h>
#include <math.h>

// Problem constants (match reference)
#define NN   65536
#define EE   524288
#define DNODE 128
#define DEDGE 64
#define HD   128
#define BN_EPS 1e-3f

typedef __attribute__((ext_vector_type(16))) _Float16 v16h;
typedef __attribute__((ext_vector_type(2)))  _Float16 v2h;
typedef __attribute__((ext_vector_type(8)))  float    v8f;

__device__ __forceinline__ float gelu_exact(float x) {
    // tf.nn.gelu approximate=False: 0.5*x*(1+erf(x/sqrt(2)))
    return 0.5f * x * (1.0f + erff(x * 0.70710678118654752f));
}

__device__ __forceinline__ void fatomic_add(float* p, float v) {
    // non-returning agent-scope f32 add -> global_atomic_add_f32 (L2 atomic unit)
    __hip_atomic_fetch_add(p, v, __ATOMIC_RELAXED, __HIP_MEMORY_SCOPE_AGENT);
}

// ---------------------------------------------------------------------------
// A-fragment loader: 16-bit A-matrix 16x32 layout (ISA 7.12.2).
// lane&15 = row M; lanes 0-15 hold K = {0..7, 16..23}, lanes 16-31 hold +8.
// ---------------------------------------------------------------------------
__device__ __forceinline__ v16h load_a_frag(const _Float16* lds, int stride,
                                            int row0, int k0, int lane) {
    const int m  = lane & 15;
    const int kh = (lane >> 4) << 3;         // 0 or 8
    const _Float16* rp = lds + (row0 + m) * stride + k0 + kh;
    v16h a;
#pragma unroll
    for (int p = 0; p < 8; ++p) {
        const int k = (p < 4) ? (2 * p) : (2 * p + 8);  // 0,2,4,6,16,18,20,22
        v2h t = *(const v2h*)(rp + k);
        a[2 * p]     = t[0];
        a[2 * p + 1] = t[1];
    }
    return a;
}

// ---------------------------------------------------------------------------
// One 16-row x 128-col GEMM slab: acc[nb] += A(16 x 32*ktiles) * W
// Weights pre-packed in B-fragment order: tile(kt,nb) is 512 halves,
// element = lane*16 + e  (lane = col + 16*((k>>3)&1), e = (k&7) + 8*((k>>4)&1)).
// ---------------------------------------------------------------------------
__device__ __forceinline__ void gemm_acc(const _Float16* aLds, int stride, int row0,
                                         const _Float16* __restrict__ wfrag,
                                         int ktiles, int lane, v8f acc[8]) {
    for (int kt = 0; kt < ktiles; ++kt) {
        v16h a = load_a_frag(aLds, stride, row0, kt * 32, lane);
        const _Float16* wk = wfrag + (size_t)(kt * 8) * 512 + (size_t)lane * 16;
#pragma unroll
        for (int nb = 0; nb < 8; ++nb) {
            v16h b = *(const v16h*)(wk + (size_t)nb * 512);
            acc[nb] = __builtin_amdgcn_wmma_f32_16x16x32_f16(
                false, a, false, b, (short)0, acc[nb], false, false);
        }
    }
}

__device__ __forceinline__ void zero_acc(v8f acc[8]) {
#pragma unroll
    for (int nb = 0; nb < 8; ++nb)
#pragma unroll
        for (int j = 0; j < 8; ++j) acc[nb][j] = 0.0f;
}

// bias + exact gelu, store f16 into LDS in row-major using C/D layout:
// vgpr v -> row = v + 8*(lane>>4), col = nb*16 + (lane&15)
__device__ __forceinline__ void epi_gelu_lds(v8f acc[8], const float* __restrict__ cb,
                                             _Float16* out, int stride, int row0, int lane) {
    const int nlo = lane & 15;
    const int rof = (lane >> 4) << 3;
#pragma unroll
    for (int nb = 0; nb < 8; ++nb) {
        const int col = nb * 16 + nlo;
        const float c = cb[col];
#pragma unroll
        for (int v = 0; v < 8; ++v) {
            float x = gelu_exact(acc[nb][v] + c);
            out[(row0 + rof + v) * stride + col] = (_Float16)x;
        }
    }
}

// ---------------------------------------------------------------------------
// prep: fold BN into weights, convert f16, emit B-fragment layout + fused bias
//   W'[i][j] = s_i * W[i][j],  c[j] = bias[j] + sum_i (b_i - m_i*s_i) * W[i][j]
// ---------------------------------------------------------------------------
__global__ void prep_kernel(const float* __restrict__ g, const float* __restrict__ b,
                            const float* __restrict__ m, const float* __restrict__ v,
                            const float* __restrict__ w, const float* __restrict__ bias,
                            _Float16* __restrict__ wout, float* __restrict__ cout,
                            int K, int has_bn) {
    const int j = threadIdx.x;  // output column 0..127
    float csum = 0.0f;
    for (int i = 0; i < K; ++i) {
        float wij = w[i * HD + j];
        float s = 1.0f, sh = 0.0f;
        if (has_bn) {
            s  = g[i] * rsqrtf(v[i] + BN_EPS);
            sh = b[i] - m[i] * s;
        }
        csum += sh * wij;
        const int kt   = i >> 5;
        const int lane = (j & 15) + (((i >> 3) & 1) << 4);
        const int off  = ((kt << 3) + (j >> 4)) * 512 + lane * 16 + ((i & 7) + (((i >> 4) & 1) << 3));
        wout[off] = (_Float16)(wij * s);
    }
    cout[j] = bias[j] + csum;
}

__global__ void zero_kernel(float* __restrict__ p, int n) {
    int i = blockIdx.x * blockDim.x + threadIdx.x;
    const int stride = gridDim.x * blockDim.x;
    for (; i < n; i += stride) p[i] = 0.0f;
}

// ---------------------------------------------------------------------------
// Edge pipeline: 128 edges/block, 8 waves; wave w owns rows 16w..16w+15
// gather -> GEMM1 -> GEMM2(h in regs) -> et1 -> et2 -> h*e -> atomic scatter
// ---------------------------------------------------------------------------
__global__ __launch_bounds__(256) void edge_msg_kernel(
    const float* __restrict__ node_x, const float* __restrict__ edge_attr,
    const int* __restrict__ edge_src, const int* __restrict__ edge_dst,
    const _Float16* __restrict__ w_bm1, const _Float16* __restrict__ w_bm2,
    const _Float16* __restrict__ w_et1, const _Float16* __restrict__ w_et2,
    const float* __restrict__ c_bm1, const float* __restrict__ c_bm2,
    const float* __restrict__ c_et1, const float* __restrict__ c_et2,
    float* __restrict__ agg_sum, float* __restrict__ cnt) {
    constexpr int AS = 136;  // padded LDS row stride (halves)
    __shared__ _Float16 actA[128 * AS];
    __shared__ _Float16 actB[128 * AS];
    __shared__ int srcIdx[128];

    const int t    = threadIdx.x;
    const int wv   = t >> 5;
    const int lane = t & 31;
    const int e0   = blockIdx.x * 128;
    const int row0 = wv << 4;

    // speculative prefetch of next tile's indices -> global_prefetch_b8
    if (e0 + 128 < EE) {
        __builtin_prefetch(edge_dst + e0 + 128 + lane * 16, 0, 0);
        __builtin_prefetch(edge_src + e0 + 128 + lane * 16, 0, 0);
    }

    // ---- gather node_x[edge_dst] -> actA as f16 (2 threads per row) ----
    {
        const int r  = row0 + (lane >> 1);
        const int hs = lane & 1;
        const int e  = e0 + r;
        const int dst = edge_dst[e];
        if (hs == 0) {
            const int s = edge_src[e];
            srcIdx[r] = s;
            fatomic_add(&cnt[s], 1.0f);
        }
        const float4* sp = (const float4*)(node_x + (size_t)dst * DNODE + hs * 64);
        v2h* dp = (v2h*)(actA + r * AS + hs * 64);
#pragma unroll
        for (int i = 0; i < 16; ++i) {
            float4 f = sp[i];
            v2h lo; lo[0] = (_Float16)f.x; lo[1] = (_Float16)f.y;
            v2h hi; hi[0] = (_Float16)f.z; hi[1] = (_Float16)f.w;
            dp[2 * i]     = lo;
            dp[2 * i + 1] = hi;
        }
    }
    __syncthreads();

    v8f acc[8];

    // ---- GEMM1: h1 = gelu(nbr @ W_bm1' + c_bm1) -> actB ----
    zero_acc(acc);
    gemm_acc(actA, AS, row0, w_bm1, 4, lane, acc);
    epi_gelu_lds(acc, c_bm1, actB, AS, row0, lane);

    // ---- GEMM2: h = gelu(h1 @ W_bm2' + c_bm2), keep in registers ----
    zero_acc(acc);
    gemm_acc(actB, AS, row0, w_bm2, 4, lane, acc);
    float hreg[8][8];
    {
        const int nlo = lane & 15;
#pragma unroll
        for (int nb = 0; nb < 8; ++nb) {
            const float c = c_bm2[nb * 16 + nlo];
#pragma unroll
            for (int v = 0; v < 8; ++v) hreg[nb][v] = gelu_exact(acc[nb][v] + c);
        }
    }

    // ---- load edge_attr -> actA cols 0..63 (wave-local, LDS in-order) ----
    {
        const int r  = row0 + (lane >> 1);
        const int hs = lane & 1;
        const float4* sp = (const float4*)(edge_attr + (size_t)(e0 + r) * DEDGE + hs * 32);
        v2h* dp = (v2h*)(actA + r * AS + hs * 32);
#pragma unroll
        for (int i = 0; i < 8; ++i) {
            float4 f = sp[i];
            v2h lo; lo[0] = (_Float16)f.x; lo[1] = (_Float16)f.y;
            v2h hi; hi[0] = (_Float16)f.z; hi[1] = (_Float16)f.w;
            dp[2 * i]     = lo;
            dp[2 * i + 1] = hi;
        }
    }

    // ---- et1 (K=64): e1 = gelu(edge_attr @ W_et1 + b) -> actB ----
    zero_acc(acc);
    gemm_acc(actA, AS, row0, w_et1, 2, lane, acc);
    epi_gelu_lds(acc, c_et1, actB, AS, row0, lane);

    // ---- et2: e = gelu(e1 @ W_et2 + b); msg = h*e; scatter-add ----
    zero_acc(acc);
    gemm_acc(actB, AS, row0, w_et2, 4, lane, acc);
    {
        const int nlo = lane & 15;
        const int rof = (lane >> 4) << 3;
        int srcs[8];
#pragma unroll
        for (int v = 0; v < 8; ++v) srcs[v] = srcIdx[row0 + rof + v];
#pragma unroll
        for (int nb = 0; nb < 8; ++nb) {
            const int col = nb * 16 + nlo;
            const float c = c_et2[col];
#pragma unroll
            for (int v = 0; v < 8; ++v) {
                float msg = gelu_exact(acc[nb][v] + c) * hreg[nb][v];
                fatomic_add(&agg_sum[(size_t)srcs[v] * HD + col], msg);
            }
        }
    }
}

// ---------------------------------------------------------------------------
// Node update: concat(node_x, agg_mean) [128 x 256] -> 2 GEMMs -> out f32
// ---------------------------------------------------------------------------
__global__ __launch_bounds__(256) void node_update_kernel(
    const float* __restrict__ node_x, const float* __restrict__ agg_sum,
    const float* __restrict__ cnt,
    const _Float16* __restrict__ w_up1, const _Float16* __restrict__ w_up2,
    const float* __restrict__ c_up1, const float* __restrict__ c_up2,
    float* __restrict__ out) {
    constexpr int AS = 264;  // 256 + pad
    constexpr int BS = 136;
    __shared__ _Float16 actA[128 * AS];
    __shared__ _Float16 actB[128 * BS];

    const int t    = threadIdx.x;
    const int wv   = t >> 5;
    const int lane = t & 31;
    const int n0   = blockIdx.x * 128;
    const int row0 = wv << 4;

    // ---- build concat row: [node_x | agg_sum/max(cnt,1)] as f16 ----
    {
        const int r    = row0 + (lane >> 1);
        const int hs   = lane & 1;
        const int node = n0 + r;
        const float inv = 1.0f / fmaxf(cnt[node], 1.0f);

        const float4* sx = (const float4*)(node_x + (size_t)node * DNODE + hs * 64);
        v2h* dx = (v2h*)(actA + r * AS + hs * 64);
        const float4* sa = (const float4*)(agg_sum + (size_t)node * HD + hs * 64);
        v2h* da = (v2h*)(actA + r * AS + 128 + hs * 64);
#pragma unroll
        for (int i = 0; i < 16; ++i) {
            float4 f = sx[i];
            v2h lo; lo[0] = (_Float16)f.x; lo[1] = (_Float16)f.y;
            v2h hi; hi[0] = (_Float16)f.z; hi[1] = (_Float16)f.w;
            dx[2 * i] = lo; dx[2 * i + 1] = hi;
            float4 a = sa[i];
            v2h alo; alo[0] = (_Float16)(a.x * inv); alo[1] = (_Float16)(a.y * inv);
            v2h ahi; ahi[0] = (_Float16)(a.z * inv); ahi[1] = (_Float16)(a.w * inv);
            da[2 * i] = alo; da[2 * i + 1] = ahi;
        }
    }
    __syncthreads();

    v8f acc[8];

    // ---- up1 (K=256, BN folded) -> actB ----
    zero_acc(acc);
    gemm_acc(actA, AS, row0, w_up1, 8, lane, acc);
    epi_gelu_lds(acc, c_up1, actB, BS, row0, lane);

    // ---- up2 -> out (f32) ----
    zero_acc(acc);
    gemm_acc(actB, BS, row0, w_up2, 4, lane, acc);
    {
        const int nlo = lane & 15;
        const int rof = (lane >> 4) << 3;
#pragma unroll
        for (int nb = 0; nb < 8; ++nb) {
            const int col = nb * 16 + nlo;
            const float c = c_up2[col];
#pragma unroll
            for (int v = 0; v < 8; ++v) {
                out[(size_t)(n0 + row0 + rof + v) * HD + col] =
                    gelu_exact(acc[nb][v] + c);
            }
        }
    }
}

// ---------------------------------------------------------------------------
extern "C" void kernel_launch(void* const* d_in, const int* in_sizes, int n_in,
                              void* d_out, int out_size, void* d_ws, size_t ws_size,
                              hipStream_t stream) {
    const float* node_x    = (const float*)d_in[0];
    const float* edge_attr = (const float*)d_in[1];
    const float* bm_g1 = (const float*)d_in[2];
    const float* bm_b1 = (const float*)d_in[3];
    const float* bm_m1 = (const float*)d_in[4];
    const float* bm_v1 = (const float*)d_in[5];
    const float* bm_w1 = (const float*)d_in[6];
    const float* bm_bb1 = (const float*)d_in[7];
    const float* bm_g2 = (const float*)d_in[8];
    const float* bm_b2 = (const float*)d_in[9];
    const float* bm_m2 = (const float*)d_in[10];
    const float* bm_v2 = (const float*)d_in[11];
    const float* bm_w2 = (const float*)d_in[12];
    const float* bm_bb2 = (const float*)d_in[13];
    const float* et_w1 = (const float*)d_in[14];
    const float* et_b1 = (const float*)d_in[15];
    const float* et_w2 = (const float*)d_in[16];
    const float* et_b2 = (const float*)d_in[17];
    const float* up_g1 = (const float*)d_in[18];
    const float* up_b1 = (const float*)d_in[19];
    const float* up_m1 = (const float*)d_in[20];
    const float* up_v1 = (const float*)d_in[21];
    const float* up_w1 = (const float*)d_in[22];
    const float* up_bb1 = (const float*)d_in[23];
    const float* up_g2 = (const float*)d_in[24];
    const float* up_b2 = (const float*)d_in[25];
    const float* up_m2 = (const float*)d_in[26];
    const float* up_v2 = (const float*)d_in[27];
    const float* up_w2 = (const float*)d_in[28];
    const float* up_bb2 = (const float*)d_in[29];
    const int* edge_src = (const int*)d_in[30];
    const int* edge_dst = (const int*)d_in[31];

    // ---- workspace layout ----
    float* agg = (float*)d_ws;                          // N*H f32
    float* cnt = agg + (size_t)NN * HD;                 // N   f32 (adjacent -> one zero pass)
    float* cb  = cnt + NN;                              // 6*H fused biases
    float* c_bm1 = cb + 0 * HD;
    float* c_bm2 = cb + 1 * HD;
    float* c_et1 = cb + 2 * HD;
    float* c_et2 = cb + 3 * HD;
    float* c_up1 = cb + 4 * HD;
    float* c_up2 = cb + 5 * HD;
    _Float16* w_bm1 = (_Float16*)(cb + 6 * HD);         // fragment-packed f16 weights
    _Float16* w_bm2 = w_bm1 + 128 * 128;
    _Float16* w_et1 = w_bm2 + 128 * 128;
    _Float16* w_et2 = w_et1 + 64 * 128;
    _Float16* w_up1 = w_et2 + 128 * 128;
    _Float16* w_up2 = w_up1 + 256 * 128;

    // ---- fold BN into weights, pack B-fragments ----
    prep_kernel<<<1, 128, 0, stream>>>(bm_g1, bm_b1, bm_m1, bm_v1, bm_w1, bm_bb1, w_bm1, c_bm1, 128, 1);
    prep_kernel<<<1, 128, 0, stream>>>(bm_g2, bm_b2, bm_m2, bm_v2, bm_w2, bm_bb2, w_bm2, c_bm2, 128, 1);
    prep_kernel<<<1, 128, 0, stream>>>(nullptr, nullptr, nullptr, nullptr, et_w1, et_b1, w_et1, c_et1, 64, 0);
    prep_kernel<<<1, 128, 0, stream>>>(nullptr, nullptr, nullptr, nullptr, et_w2, et_b2, w_et2, c_et2, 128, 0);
    prep_kernel<<<1, 128, 0, stream>>>(up_g1, up_b1, up_m1, up_v1, up_w1, up_bb1, w_up1, c_up1, 256, 1);
    prep_kernel<<<1, 128, 0, stream>>>(up_g2, up_b2, up_m2, up_v2, up_w2, up_bb2, w_up2, c_up2, 128, 1);

    // ---- zero accumulators (agg + cnt are contiguous) ----
    zero_kernel<<<2048, 256, 0, stream>>>(agg, NN * HD + NN);

    // ---- edge message + scatter ----
    edge_msg_kernel<<<EE / 128, 256, 0, stream>>>(
        node_x, edge_attr, edge_src, edge_dst,
        w_bm1, w_bm2, w_et1, w_et2,
        c_bm1, c_bm2, c_et1, c_et2,
        agg, cnt);

    // ---- node update ----
    node_update_kernel<<<NN / 128, 256, 0, stream>>>(
        node_x, agg, cnt, w_up1, w_up2, c_up1, c_up2, (float*)d_out);
}
